// CrossCovarianceAttention_68135361184436
// MI455X (gfx1250) — matrix operand order repaired and checked by reading, not compiled, not run on hardware
//
#include <hip/hip_runtime.h>

typedef __attribute__((ext_vector_type(16))) __bf16 v16bf;
typedef __attribute__((ext_vector_type(8)))  float  v8f;

#define EPS 1e-12f
#define B   16
#define C   384
#define HW  2304   // 48*48
#define NH  8
#define HD  48
#define C3  1152

__device__ __forceinline__ __bf16 f2bf(float f) { return (__bf16)f; }

// ---------------- CDNA5 async-copy helpers ----------------
// GLOBAL_LOAD_ASYNC_TO_LDS_B128: per-lane 16B global->LDS, tracked by ASYNCcnt.
__device__ __forceinline__ void async_b128(const __bf16* g, __bf16* l) {
  unsigned lds = (unsigned)(uintptr_t)l;   // flat addr[31:0] == LDS byte address
  asm volatile("global_load_async_to_lds_b128 %0, %1, off"
               :: "v"(lds), "v"(g) : "memory");
}
__device__ __forceinline__ void wait_async_le4() {
  asm volatile("s_wait_asynccnt 0x4" ::: "memory");
}
__device__ __forceinline__ void wait_async_0() {
  asm volatile("s_wait_asynccnt 0x0" ::: "memory");
}

// ---------------- WMMA operand fetch ----------------
// A frag from LDS row (row-major [16 x 32] tile): lane L holds row M=L&15,
// kbase=(L<16)?0:16 ; elements 0..15 = K kb..kb+15.
__device__ __forceinline__ v16bf lds_a_frag(const __bf16* sArow, int lane) {
  int kb = (lane < 16) ? 0 : 16;
  const __bf16* p0 = sArow + kb;
  const __bf16* p1 = sArow + kb + 8;
  v16bf a;
#pragma unroll
  for (int i = 0; i < 8; ++i) { a[i] = p0[i]; a[8 + i] = p1[i]; }
  return a;
}
// B frag from transposed LDS row BT[N][K]: lane L holds col N=L&15,
// lanes 0-15: K=0..15, lanes 16-31: K=16..31.
__device__ __forceinline__ v16bf lds_b_frag(const __bf16* sBrow, int lane) {
  int kb = (lane < 16) ? 0 : 16;
  const __bf16* p = sBrow + kb;
  v16bf b;
#pragma unroll
  for (int i = 0; i < 16; ++i) b[i] = p[i];
  return b;
}
// B frag straight from global transposed storage BT[N][K] stride ldb.
__device__ __forceinline__ v16bf load_bt_tile(const __bf16* BTbase, int ldb, int lane) {
  int n  = lane & 15;
  int kb = (lane < 16) ? 0 : 16;
  const __bf16* row = BTbase + (size_t)n * ldb;
  v16bf b;
#pragma unroll
  for (int j = 0; j < 16; ++j) b[j] = row[kb + j];
  return b;
}

// Issue one double-buffer chunk: A tile 64x32 + B tile 64x32 (bf16), 128 threads,
// 2+2 b128 async instructions per thread (4 per wave -> ASYNCcnt += 4).
__device__ __forceinline__ void async_tile_load(const __bf16* gA, int lda,
                                                const __bf16* gB, int ldb,
                                                __bf16 (*sA)[32], __bf16 (*sB)[32],
                                                int tid) {
#pragma unroll
  for (int r = 0; r < 2; ++r) {
    int n = tid + r * 128;          // 256 chunks of 16B
    int row = n >> 2, co = (n & 3) * 8;
    async_b128(gA + (size_t)row * lda + co, &sA[row][co]);
  }
#pragma unroll
  for (int r = 0; r < 2; ++r) {
    int n = tid + r * 128;
    int row = n >> 2, co = (n & 3) * 8;
    async_b128(gB + (size_t)row * ldb + co, &sB[row][co]);
  }
}

// ---------------- prep: bf16 conversion + transposes ----------------
__global__ void prep_weights(const float* __restrict__ w_qkv, const float* __restrict__ w_out,
                             __bf16* __restrict__ wqkvT, __bf16* __restrict__ woutT) {
  int idx = blockIdx.x * blockDim.x + threadIdx.x;
  int total1 = C3 * C;
  if (idx < total1) {
    int n = idx / C, k = idx % C;
    wqkvT[(size_t)n * C + k] = f2bf(w_qkv[(size_t)k * C3 + n]);
  } else if (idx < total1 + C * C) {
    int t = idx - total1;
    int n = t / C, k = t % C;
    woutT[(size_t)n * C + k] = f2bf(w_out[(size_t)k * C + n]);
  }
}

// x [b, c, n] -> xt bf16 [b, n, c], 32x32 LDS-tiled transpose
__global__ void __launch_bounds__(256)
prep_x(const float* __restrict__ x, __bf16* __restrict__ xt) {
  __shared__ float tile[32][33];
  int b = blockIdx.x, ct = blockIdx.y, nt = blockIdx.z;
  int tx = threadIdx.x & 31, ty = threadIdx.x >> 5;   // 32 x 8
#pragma unroll
  for (int r = 0; r < 4; ++r) {
    int c = ct * 32 + ty + r * 8;
    tile[ty + r * 8][tx] = x[((size_t)b * C + c) * HW + nt * 32 + tx];
  }
  __syncthreads();
#pragma unroll
  for (int r = 0; r < 4; ++r) {
    int n = nt * 32 + ty + r * 8;
    xt[((size_t)b * HW + n) * C + ct * 32 + tx] = f2bf(tile[tx][ty + r * 8]);
  }
}

// ---------------- K1: qkv = xt @ w_qkv  (async-LDS double-buffered) ----------------
__global__ void __launch_bounds__(128)
qkv_gemm(const __bf16* __restrict__ xt, const __bf16* __restrict__ wqkvT,
         __bf16* __restrict__ qkv) {
  __shared__ __align__(16) __bf16 sA[2][64][32];
  __shared__ __align__(16) __bf16 sB[2][64][32];
  int tid = threadIdx.x;
  int lane = tid & 31, wave = tid >> 5;
  int b = blockIdx.x, mt = blockIdx.y, ns = blockIdx.z;
  int m0 = mt * 64, n0 = ns * 64;
  const __bf16* Abase = xt + ((size_t)b * HW + m0) * C;   // lda = C
  const __bf16* Bbase = wqkvT + (size_t)n0 * C;           // ldb = C
  v8f acc[4] = {};
  const int KCH = C / 32;  // 12
  async_tile_load(Abase, C, Bbase, C, sA[0], sB[0], tid);
  for (int i = 0; i < KCH; ++i) {
    int cur = i & 1;
    if (i + 1 < KCH) {
      __syncthreads();   // all waves done reading buf[cur^1]
      async_tile_load(Abase + (i + 1) * 32, C, Bbase + (i + 1) * 32, C,
                      sA[cur ^ 1], sB[cur ^ 1], tid);
      wait_async_le4();  // chunk i (this wave) complete
    } else {
      wait_async_0();
    }
    __syncthreads();     // chunk i visible to all waves
    v16bf a = lds_a_frag(&sA[cur][wave * 16 + (lane & 15)][0], lane);
#pragma unroll
    for (int t = 0; t < 4; ++t) {
      v16bf bm = lds_b_frag(&sB[cur][t * 16 + (lane & 15)][0], lane);
      acc[t] = __builtin_amdgcn_wmma_f32_16x16x32_bf16(false, a, false, bm,
                                                       (short)0, acc[t], false, false);
    }
  }
  int col = lane & 15;
  int rb  = (lane < 16) ? 0 : 8;
#pragma unroll
  for (int t = 0; t < 4; ++t)
#pragma unroll
    for (int j = 0; j < 8; ++j) {
      int row = m0 + wave * 16 + rb + j;
      qkv[((size_t)b * HW + row) * C3 + n0 + t * 16 + col] = f2bf(acc[t][j]);
    }
}

// ---------------- K2: per-channel L2 norms over tokens ----------------
__global__ void __launch_bounds__(256)
col_norms(const __bf16* __restrict__ qkv, const float* __restrict__ temperature,
          float* __restrict__ invq, float* __restrict__ invk) {
  int b  = blockIdx.x;
  int ch = blockIdx.y;          // 0..767 : <C => q channel, else k channel
  bool is_k = ch >= C;
  int cc  = is_k ? (ch - C) : ch;
  int off = is_k ? C : 0;
  float s = 0.f;
  for (int n = threadIdx.x; n < HW; n += 256) {
    float v = (float)qkv[((size_t)b * HW + n) * C3 + off + cc];
    s += v * v;
  }
  __shared__ float red[256];
  red[threadIdx.x] = s;
  __syncthreads();
  for (int st = 128; st > 0; st >>= 1) {
    if (threadIdx.x < st) red[threadIdx.x] += red[threadIdx.x + st];
    __syncthreads();
  }
  if (threadIdx.x == 0) {
    float inv = 1.f / fmaxf(sqrtf(red[0]), EPS);
    if (is_k) invk[b * C + cc] = inv * temperature[cc / HD];
    else      invq[b * C + cc] = inv;
  }
}

// ---------------- K3: attn = softmax( (invk*k)^T (invq*q) ) ----------------
__global__ void __launch_bounds__(32)
attn_kernel(const __bf16* __restrict__ qkv, const float* __restrict__ invq,
            const float* __restrict__ invk, __bf16* __restrict__ attn_s) {
  int lane = threadIdx.x;
  int b = blockIdx.x, h = blockIdx.y;
  const __bf16* qb = qkv + (size_t)b * HW * C3 + h * HD;       // q columns
  const __bf16* kb = qkv + (size_t)b * HW * C3 + C + h * HD;   // k columns
  v8f acc[3][3] = {};
  int m   = lane & 15;
  int akb = (lane < 16) ? 0 : 16;
  int nn  = lane & 15;
  int bkb = (lane < 16) ? 0 : 16;
  for (int kt = 0; kt < HW; kt += 32) {
    v16bf a[3], bm[3];
#pragma unroll
    for (int t = 0; t < 3; ++t) {
#pragma unroll
      for (int j = 0; j < 8; ++j) {   // A[p,n] = k[n, t*16+m] (gather over tokens)
        int kk = akb + 2 * j;
        a[t][2 * j]     = kb[(size_t)(kt + kk)     * C3 + t * 16 + m];
        a[t][2 * j + 1] = kb[(size_t)(kt + kk + 1) * C3 + t * 16 + m];
      }
#pragma unroll
      for (int j = 0; j < 8; ++j) {   // B[n,q'] = q[n, t*16+nn]
        bm[t][2 * j]     = qb[(size_t)(kt + bkb + 2 * j)     * C3 + t * 16 + nn];
        bm[t][2 * j + 1] = qb[(size_t)(kt + bkb + 2 * j + 1) * C3 + t * 16 + nn];
      }
    }
#pragma unroll
    for (int pt = 0; pt < 3; ++pt)
#pragma unroll
      for (int qt = 0; qt < 3; ++qt)
        acc[pt][qt] = __builtin_amdgcn_wmma_f32_16x16x32_bf16(false, a[pt], false, bm[qt],
                                                              (short)0, acc[pt][qt], false, false);
  }
  __shared__ float att[HD][HD + 1];
  int col = lane & 15;
  int rb  = (lane < 16) ? 0 : 8;
#pragma unroll
  for (int pt = 0; pt < 3; ++pt)
#pragma unroll
    for (int qt = 0; qt < 3; ++qt)
#pragma unroll
      for (int j = 0; j < 8; ++j) {
        int p  = pt * 16 + rb + j;
        int qi = qt * 16 + col;
        att[p][qi] = acc[pt][qt][j] * invk[b * C + h * HD + p] * invq[b * C + h * HD + qi];
      }
  __syncthreads();
  // softmax over q' per row p; store transposed [q'][p], p padded to 64
  __bf16* out = attn_s + (size_t)(b * NH + h) * HD * 64;
  for (int p = lane; p < HD; p += 32) {
    float mx = -1e30f;
    for (int qi = 0; qi < HD; ++qi) mx = fmaxf(mx, att[p][qi]);
    float sum = 0.f;
    for (int qi = 0; qi < HD; ++qi) sum += __expf(att[p][qi] - mx);
    float r = 1.f / sum;
    for (int qi = 0; qi < HD; ++qi) out[qi * 64 + p] = f2bf(__expf(att[p][qi] - mx) * r);
  }
  for (int idx = lane; idx < HD * 16; idx += 32) {  // zero K-pad p=48..63
    int qi = idx / 16, p = HD + (idx % 16);
    out[qi * 64 + p] = f2bf(0.f);
  }
}

// ---------------- K4: out_pre = v @ attn ----------------
__global__ void __launch_bounds__(32)
v_attn(const __bf16* __restrict__ qkv, const __bf16* __restrict__ attn_s,
       __bf16* __restrict__ out_pre) {
  int lane = threadIdx.x;
  int b = blockIdx.x, h = blockIdx.y, mt = blockIdx.z;
  int m0 = mt * 16;
  const __bf16* vb = qkv + ((size_t)b * HW + m0) * C3 + 2 * C + h * HD;  // v[n,p]
  const __bf16* BT = attn_s + (size_t)(b * NH + h) * HD * 64;
  v8f acc[3] = {};
  int m   = lane & 15;
  int akb = (lane < 16) ? 0 : 16;
#pragma unroll
  for (int kt = 0; kt < 64; kt += 32) {
    v16bf a;
#pragma unroll
    for (int j = 0; j < 8; ++j) {
      int kk = kt + akb + 2 * j;
      a[2 * j]     = (kk     < HD) ? vb[(size_t)m * C3 + kk]     : f2bf(0.f);
      a[2 * j + 1] = (kk + 1 < HD) ? vb[(size_t)m * C3 + kk + 1] : f2bf(0.f);
    }
#pragma unroll
    for (int t = 0; t < 3; ++t) {
      v16bf bm = load_bt_tile(BT + (size_t)t * 16 * 64 + kt, 64, lane);
      acc[t] = __builtin_amdgcn_wmma_f32_16x16x32_bf16(false, a, false, bm,
                                                       (short)0, acc[t], false, false);
    }
  }
  int col = lane & 15;
  int rb  = (lane < 16) ? 0 : 8;
#pragma unroll
  for (int t = 0; t < 3; ++t)
#pragma unroll
    for (int j = 0; j < 8; ++j) {
      int row = m0 + rb + j;
      out_pre[((size_t)b * HW + row) * C + h * HD + t * 16 + col] = f2bf(acc[t][j]);
    }
}

// ---------------- K5: out = out_pre @ w_out + b_out (async-LDS double-buffered) ----------------
__global__ void __launch_bounds__(128)
out_proj(const __bf16* __restrict__ out_pre, const __bf16* __restrict__ woutT,
         const float* __restrict__ b_out, float* __restrict__ out) {
  __shared__ __align__(16) __bf16 sA[2][64][32];
  __shared__ __align__(16) __bf16 sB[2][64][32];
  int tid = threadIdx.x;
  int lane = tid & 31, wave = tid >> 5;
  int b = blockIdx.x, mt = blockIdx.y, ns = blockIdx.z;
  int m0 = mt * 64, n0 = ns * 64;
  const __bf16* Abase = out_pre + ((size_t)b * HW + m0) * C;
  const __bf16* Bbase = woutT + (size_t)n0 * C;
  v8f acc[4] = {};
  const int KCH = C / 32;  // 12
  async_tile_load(Abase, C, Bbase, C, sA[0], sB[0], tid);
  for (int i = 0; i < KCH; ++i) {
    int cur = i & 1;
    if (i + 1 < KCH) {
      __syncthreads();
      async_tile_load(Abase + (i + 1) * 32, C, Bbase + (i + 1) * 32, C,
                      sA[cur ^ 1], sB[cur ^ 1], tid);
      wait_async_le4();
    } else {
      wait_async_0();
    }
    __syncthreads();
    v16bf a = lds_a_frag(&sA[cur][wave * 16 + (lane & 15)][0], lane);
#pragma unroll
    for (int t = 0; t < 4; ++t) {
      v16bf bm = lds_b_frag(&sB[cur][t * 16 + (lane & 15)][0], lane);
      acc[t] = __builtin_amdgcn_wmma_f32_16x16x32_bf16(false, a, false, bm,
                                                       (short)0, acc[t], false, false);
    }
  }
  int col = lane & 15;
  int rb  = (lane < 16) ? 0 : 8;
#pragma unroll
  for (int t = 0; t < 4; ++t) {
    int cc = n0 + t * 16 + col;
    float bias = b_out[cc];
#pragma unroll
    for (int j = 0; j < 8; ++j) {
      int row = m0 + wave * 16 + rb + j;
      out[((size_t)b * C + cc) * HW + row] = acc[t][j] + bias;
    }
  }
}

extern "C" void kernel_launch(void* const* d_in, const int* in_sizes, int n_in,
                              void* d_out, int out_size, void* d_ws, size_t ws_size,
                              hipStream_t stream) {
  const float* x     = (const float*)d_in[0];
  const float* w_qkv = (const float*)d_in[1];
  const float* temp  = (const float*)d_in[2];
  const float* w_out = (const float*)d_in[3];
  const float* b_out = (const float*)d_in[4];
  (void)in_sizes; (void)n_in; (void)out_size; (void)ws_size;

  char* ws = (char*)d_ws;
  size_t off = 0;
  auto alloc = [&](size_t bytes) -> void* {
    void* p = ws + off;
    off = (off + bytes + 255) & ~(size_t)255;
    return p;
  };
  __bf16* wqkvT = (__bf16*)alloc((size_t)C3 * C * 2);
  __bf16* woutT = (__bf16*)alloc((size_t)C * C * 2);
  __bf16* xt    = (__bf16*)alloc((size_t)B * HW * C * 2);   // reused as out_pre
  __bf16* qkv   = (__bf16*)alloc((size_t)B * HW * C3 * 2);
  float*  invq  = (float*) alloc((size_t)B * C * 4);
  float*  invk  = (float*) alloc((size_t)B * C * 4);
  __bf16* attn  = (__bf16*)alloc((size_t)B * NH * HD * 64 * 2);

  {
    int tot = C3 * C + C * C;
    prep_weights<<<(tot + 255) / 256, 256, 0, stream>>>(w_qkv, w_out, wqkvT, woutT);
  }
  prep_x<<<dim3(B, C / 32, HW / 32), 256, 0, stream>>>(x, xt);
  qkv_gemm<<<dim3(B, HW / 64, C3 / 64), 128, 0, stream>>>(xt, wqkvT, qkv);
  col_norms<<<dim3(B, 2 * C), 256, 0, stream>>>(qkv, temp, invq, invk);
  attn_kernel<<<dim3(B, NH), 32, 0, stream>>>(qkv, invq, invk, attn);
  v_attn<<<dim3(B, NH, HW / 16), 32, 0, stream>>>(qkv, attn, xt /* out_pre */);
  out_proj<<<dim3(B, HW / 64, C / 64), 128, 0, stream>>>(xt, woutT, b_out, (float*)d_out);
}
